// GATLayer_25039659336028
// MI455X (gfx1250) — matrix operand order repaired and checked by reading, not compiled, not run on hardware
//
#include <hip/hip_runtime.h>

// ---------------------------------------------------------------------------
// GAT layer for MI455X (gfx1250, wave32, WMMA bf16 16x16x32)
//   N=4096 nodes, F_in=F_out=256, H=4 heads, D=64
// Single-pass softmax (scores are O(1), exp cannot overflow):
//   O_i  = sum_j [adj_ij>0] * exp(lrelu(a_i+b_j)) * adj_ij * h_j
//   Z_i  = sum_j [adj_ij>0] * exp(lrelu(a_i+b_j))
//   out  = O_i / Z_i    (Z==0 -> 0, matches nan_to_num)
// ---------------------------------------------------------------------------

typedef __bf16 bf16;
typedef __attribute__((ext_vector_type(16))) __bf16 v16bf;
typedef __attribute__((ext_vector_type(8)))  __bf16 v8bf;
typedef __attribute__((ext_vector_type(8)))  float  v8f;

#define NN        4096
#define FIN       256
#define FOUT      256
#define NHEAD     4
#define HDIM      64
#define NEG_SLOPE 0.2f

#ifndef __has_builtin
#define __has_builtin(x) 0
#endif
#if __has_builtin(__builtin_amdgcn_global_load_async_to_lds_b128) && \
    __has_builtin(__builtin_amdgcn_s_wait_asynccnt)
#define USE_ASYNC_LDS 1
#else
#define USE_ASYNC_LDS 0
#endif

#if USE_ASYNC_LDS
// Builtin parameter type per hipcc diagnostic: pointer to
// "__attribute__((vector_size(4*sizeof(int)))) int" in AS(1)/AS(3).
typedef int v4i_async __attribute__((vector_size(16)));
typedef __attribute__((address_space(1))) v4i_async* gp16_t;
typedef __attribute__((address_space(3))) v4i_async* lp16_t;

// 16B global -> LDS async copy (per lane), tracked by ASYNCcnt.
__device__ __forceinline__ void async_tile16(const void* g, void* l) {
  __builtin_amdgcn_global_load_async_to_lds_b128(
      (gp16_t)((v4i_async*)g),   // bitcast, then generic->AS1 addrspacecast
      (lp16_t)((v4i_async*)l),   // bitcast, then generic->AS3 addrspacecast
      0, 0);
}
#endif

// ---------------------------------------------------------------------------
// Kernel 1: f32 -> bf16 conversion of x and W
// ---------------------------------------------------------------------------
__global__ __launch_bounds__(256) void k_convert(const float* __restrict__ x,
                                                 const float* __restrict__ W,
                                                 bf16* __restrict__ x_bf,
                                                 bf16* __restrict__ W_bf) {
  int i = blockIdx.x * blockDim.x + threadIdx.x;
  if (i < NN * FIN)   x_bf[i] = (bf16)x[i];
  if (i < FOUT * FIN) W_bf[i] = (bf16)W[i];
}

// ---------------------------------------------------------------------------
// Kernel 2: projection h = x @ W.T via v_wmma_f32_16x16x32_bf16.
// Output stored TRANSPOSED: hT[f, n]  (f-major) so that attention B-fragments
// are contiguous 32B loads.  One wave computes one 16x16 tile over K=256.
// grid (NN/16, 4), block 128 (4 waves), wave w -> col tile blockIdx.y*4+w.
// ---------------------------------------------------------------------------
__global__ __launch_bounds__(128) void k_proj(const bf16* __restrict__ x_bf,
                                              const bf16* __restrict__ W_bf,
                                              bf16* __restrict__ hT) {
  const int lane = threadIdx.x & 31;
  const int wave = threadIdx.x >> 5;
  const int g    = lane >> 4;      // lane half
  const int m    = lane & 15;
  const int n0   = blockIdx.x * 16;           // row block (nodes)
  const int ct   = blockIdx.y * 4 + wave;     // 16-col tile of F_out

  const bf16* xrow = x_bf + (size_t)(n0 + m) * FIN;      // A: row M = m
  const bf16* wrow = W_bf + (size_t)(ct * 16 + m) * FIN; // B: col N = m -> W[f,k]

  v8f c = {};
  #pragma unroll
  for (int k0 = 0; k0 < FIN; k0 += 32) {
    // A fragment (16x32 bf16): K = e + 16g  (two 16B loads)
    v8bf alo = *(const v8bf*)(xrow + k0 + 16 * g);
    v8bf ahi = *(const v8bf*)(xrow + k0 + 8 + 16 * g);
    v16bf a;
    #pragma unroll
    for (int e = 0; e < 8; ++e) { a[e] = alo[e]; a[e + 8] = ahi[e]; }
    // B fragment (32x16 bf16): K = e + 16g, contiguous in memory (32B load)
    v16bf b = *(const v16bf*)(wrow + k0 + 16 * g);
    c = __builtin_amdgcn_wmma_f32_16x16x32_bf16(false, a, false, b, (short)0, c,
                                                false, false);
  }
  // C layout: lane m holds column f=ct*16+m, rows n0 + r + 8g (r=0..7) -> one
  // contiguous 16B bf16 store into hT[f, n].
  v8bf o;
  #pragma unroll
  for (int r = 0; r < 8; ++r) o[r] = (bf16)c[r];
  *(v8bf*)(hT + (size_t)(ct * 16 + m) * NN + n0 + 8 * g) = o;
}

// ---------------------------------------------------------------------------
// Kernel 3: attn_src[n,h] = sum_d hT[h*64+d, n] * a_src[h,d]   (and dst)
// Coalesced: consecutive lanes read consecutive n of each hT row.
// ---------------------------------------------------------------------------
__global__ __launch_bounds__(128) void k_attn(const bf16* __restrict__ hT,
                                              const float* __restrict__ a_src,
                                              const float* __restrict__ a_dst,
                                              float* __restrict__ attn_src,
                                              float* __restrict__ attn_dst) {
  __shared__ float s_src[NHEAD * HDIM], s_dst[NHEAD * HDIM];
  for (int i = threadIdx.x; i < NHEAD * HDIM; i += blockDim.x) {
    s_src[i] = a_src[i];
    s_dst[i] = a_dst[i];
  }
  __syncthreads();
  const int n = blockIdx.x * blockDim.x + threadIdx.x;
  #pragma unroll
  for (int h = 0; h < NHEAD; ++h) {
    float ss = 0.f, sd = 0.f;
    for (int d = 0; d < HDIM; ++d) {
      float hv = (float)hT[(size_t)(h * HDIM + d) * NN + n];
      ss += hv * s_src[h * HDIM + d];
      sd += hv * s_dst[h * HDIM + d];
    }
    attn_src[n * NHEAD + h] = ss;
    attn_dst[n * NHEAD + h] = sd;
  }
}

// ---------------------------------------------------------------------------
// Kernel 4: fused attention + aggregation.
// Block = 128 threads = 4 waves; wave w == head w.  Each block owns 16 rows i.
// Streams j in chunks of 32: adj tile async-DMAs into LDS (shared by all
// heads; each wave issues its slice then waits its own ASYNCcnt before the
// barrier), weight tile built in registers directly in WMMA-A layout, 4 WMMAs
// per chunk per head accumulate O (16 x 64 f32), Z accumulated alongside.
// ---------------------------------------------------------------------------
__global__ __launch_bounds__(128) void k_gat(const float* __restrict__ adj,
                                             const bf16* __restrict__ hT,
                                             const float* __restrict__ attn_src,
                                             const float* __restrict__ attn_dst,
                                             float* __restrict__ out) {
  __shared__ float adj_lds[16 * 32];       // [i_local][j_local]
  __shared__ float adst_lds[32 * NHEAD];   // [j_local][head]

  const int t    = threadIdx.x;
  const int lane = t & 31;
  const int head = t >> 5;
  const int g    = lane >> 4;
  const int m    = lane & 15;
  const int i0   = blockIdx.x * 16;

  const float asrc_i = attn_src[(i0 + m) * NHEAD + head];  // row M = m

  // B-fragment base rows of hT for the 4 d-tiles of this head
  const bf16* bp0 = hT + (size_t)(head * HDIM +  0 + m) * NN + 16 * g;
  const bf16* bp1 = hT + (size_t)(head * HDIM + 16 + m) * NN + 16 * g;
  const bf16* bp2 = hT + (size_t)(head * HDIM + 32 + m) * NN + 16 * g;
  const bf16* bp3 = hT + (size_t)(head * HDIM + 48 + m) * NN + 16 * g;

  v8f c0 = {}, c1 = {}, c2 = {}, c3 = {};
  float zacc = 0.f;

  const int   arow = t >> 3;                  // 0..15
  const int   acol = (t & 7) * 4;             // 0,4,...,28
  const float* adj_src = adj + (size_t)(i0 + arow) * NN + acol;

  for (int j0 = 0; j0 < NN; j0 += 32) {
    // ---- stage adj tile (16x32 f32 = 2KB) and adst slice (32x4 f32) ----
#if USE_ASYNC_LDS
    async_tile16(adj_src + j0, &adj_lds[arow * 32 + acol]);
    if (t < 32)  // wave 0 pulls the 128-float adst slice (contiguous [j][h])
      async_tile16(attn_dst + j0 * NHEAD + t * 4, &adst_lds[t * 4]);
    __builtin_amdgcn_s_wait_asynccnt(0);  // each wave drains its own issues
#else
    *(float4*)&adj_lds[arow * 32 + acol] = *(const float4*)(adj_src + j0);
    adst_lds[t] = attn_dst[j0 * NHEAD + t];
#endif
    __syncthreads();
    if (j0 + 32 < NN)                         // pull next tile toward L2/WGP$
      __builtin_prefetch(adj_src + j0 + 32, 0, 0);

    // ---- build weight tile directly in 16-bit WMMA A layout ----
    v16bf a;
    #pragma unroll
    for (int e = 0; e < 16; ++e) {
      const int jl = e + 16 * g;  // K index
      float s  = asrc_i + adst_lds[jl * NHEAD + head];
      s        = (s >= 0.f) ? s : NEG_SLOPE * s;               // leaky ReLU
      float ex = __expf(s);
      float av = adj_lds[m * 32 + jl];
      float ok = (av > 0.f) ? 1.f : 0.f;                       // edge mask
      zacc    += ok * ex;                                      // softmax denom
      a[e]     = (bf16)(ok * ex * av);                         // numerator wt
    }

    // ---- O += W(16x32) x V(32x64) : 4 WMMAs ----
    v16bf b;
    b  = *(const v16bf*)(bp0 + j0);
    c0 = __builtin_amdgcn_wmma_f32_16x16x32_bf16(false, a, false, b, (short)0, c0, false, false);
    b  = *(const v16bf*)(bp1 + j0);
    c1 = __builtin_amdgcn_wmma_f32_16x16x32_bf16(false, a, false, b, (short)0, c1, false, false);
    b  = *(const v16bf*)(bp2 + j0);
    c2 = __builtin_amdgcn_wmma_f32_16x16x32_bf16(false, a, false, b, (short)0, c2, false, false);
    b  = *(const v16bf*)(bp3 + j0);
    c3 = __builtin_amdgcn_wmma_f32_16x16x32_bf16(false, a, false, b, (short)0, c3, false, false);
    __syncthreads();
  }

  // ---- normalize: row M is held by lanes M and M+16 ----
  float zred = zacc + __shfl_xor(zacc, 16);   // lane L -> Z[row L%16]
  float inv[8];
  #pragma unroll
  for (int r = 0; r < 8; ++r) {
    float zr = __shfl(zred, r + 8 * g);       // Z for this element's row
    inv[r]   = (zr > 0.f) ? (1.f / zr) : 0.f; // fully-masked row -> 0
  }
  #pragma unroll
  for (int r = 0; r < 8; ++r) {
    const int row = i0 + r + 8 * g;
    float* po = out + (size_t)row * (NHEAD * HDIM) + head * HDIM + m;
    po[0]  = c0[r] * inv[r];
    po[16] = c1[r] * inv[r];
    po[32] = c2[r] * inv[r];
    po[48] = c3[r] * inv[r];
  }
}

// ---------------------------------------------------------------------------
extern "C" void kernel_launch(void* const* d_in, const int* in_sizes, int n_in,
                              void* d_out, int out_size, void* d_ws,
                              size_t ws_size, hipStream_t stream) {
  (void)in_sizes; (void)n_in; (void)out_size; (void)ws_size;
  const float* x     = (const float*)d_in[0];  // [4096,256]
  const float* adj   = (const float*)d_in[1];  // [4096,4096]
  const float* W     = (const float*)d_in[2];  // [256,256]
  const float* a_src = (const float*)d_in[3];  // [4,64]
  const float* a_dst = (const float*)d_in[4];  // [4,64]
  float* out = (float*)d_out;                  // [4096,256]

  char* ws = (char*)d_ws;
  bf16* x_bf = (bf16*)ws;              ws += (size_t)NN * FIN * sizeof(bf16);
  bf16* W_bf = (bf16*)ws;              ws += (size_t)FOUT * FIN * sizeof(bf16);
  bf16* hT   = (bf16*)ws;              ws += (size_t)FOUT * NN * sizeof(bf16);
  float* asr = (float*)ws;             ws += (size_t)NN * NHEAD * sizeof(float);
  float* ads = (float*)ws;

  k_convert<<<dim3((NN * FIN) / 256), 256, 0, stream>>>(x, W, x_bf, W_bf);
  k_proj<<<dim3(NN / 16, 4), 128, 0, stream>>>(x_bf, W_bf, hT);
  k_attn<<<dim3(NN / 128), 128, 0, stream>>>(hT, a_src, a_dst, asr, ads);
  k_gat<<<dim3(NN / 16), 128, 0, stream>>>(adj, hT, asr, ads, out);
}
